// MetaEmbedding_Classifier_75067438399580
// MI455X (gfx1250) — compile-verified, matrix operand order checked by reading, not run on hardware
//
#include <hip/hip_runtime.h>
#include <hip/hip_bf16.h>
#include <math.h>

// ---------------------------------------------------------------------------
// MetaEmbedding classifier forward for MI455X (gfx1250, wave32, WMMA).
// All five GEMMs run as bf16 WMMA (v_wmma_f32_16x16x32_bf16) with f32 accum.
// Wave tile 32x64 (2x4 WMMA accumulators): 8 WMMAs per 12 b128 loads.
// ---------------------------------------------------------------------------

typedef __attribute__((ext_vector_type(8)))  __bf16 v8bf;
typedef __attribute__((ext_vector_type(16))) __bf16 v16bf;
typedef __attribute__((ext_vector_type(8)))  float  v8f;

#define FEAT   2048
#define NCLS   1000
#define NCLSP  1024
#define BATCH  8192
#define COS_SCALE   16.0f
#define REACH_SCALE 10.0f
#define EPS_R       1e-5f

// ------------------------- block reductions --------------------------------
__device__ __forceinline__ float block_reduce_sum(float v) {
  __shared__ float s[256];
  int t = threadIdx.x;
  s[t] = v; __syncthreads();
  for (int off = 128; off > 0; off >>= 1) {
    if (t < off) s[t] += s[t + off];
    __syncthreads();
  }
  float r = s[0]; __syncthreads();
  return r;
}
__device__ __forceinline__ float block_reduce_min(float v) {
  __shared__ float s[256];
  int t = threadIdx.x;
  s[t] = v; __syncthreads();
  for (int off = 128; off > 0; off >>= 1) {
    if (t < off) s[t] = fminf(s[t], s[t + off]);
    __syncthreads();
  }
  float r = s[0]; __syncthreads();
  return r;
}
__device__ __forceinline__ float block_reduce_max(float v) {
  __shared__ float s[256];
  int t = threadIdx.x;
  s[t] = v; __syncthreads();
  for (int off = 128; off > 0; off >>= 1) {
    if (t < off) s[t] = fmaxf(s[t], s[t + off]);
    __syncthreads();
  }
  float r = s[0]; __syncthreads();
  return r;
}

// --------------------- WMMA fragment loader --------------------------------
// 16-bit A-matrix 16x32 per-lane layout (ISA 7.12.2):
//   lanes 0-15  (hi=0): row M=lane,  K = {k0..k0+7,  k0+16..k0+23}
//   lanes 16-31 (hi=1): row M=l16,   K = {k0+8..15,  k0+24..31}
// B uses the same pattern with N playing the role of M (B stored [N][K] row
// major), so both operands load with two contiguous 16B reads per lane.
__device__ __forceinline__ v16bf frag_ld(const __bf16* __restrict__ row, int k0, int hi) {
  v8bf a = *reinterpret_cast<const v8bf*>(row + k0 + hi * 8);
  v8bf b = *reinterpret_cast<const v8bf*>(row + k0 + 16 + hi * 8);
  return __builtin_shufflevector(a, b, 0,1,2,3,4,5,6,7,8,9,10,11,12,13,14,15);
}

// --------------------------- GEMM template ---------------------------------
// C[M x Npad] = A[M x K](row) * B[Npad x K](row)^T , bf16 in, f32 accum.
// Block: 256 threads = 8 waves; wave tile 32x64 (2x4 wmma accs);
// block tile 256x64.
// EPI 0: store f32 acc to out0 (stride ld0)
// EPI 1: selector epilogue: s=tanh(acc+bias[col]); inf=s*memf; out1=inf;
//        out0 = reach[m]*(x[m,col]+inf)            (xo)
// EPI 2: store only col < Nreal to out0 (stride ld0)
template<int EPI>
__global__ __launch_bounds__(256) void gemm_bf16_wmma(
    const __bf16* __restrict__ A, const __bf16* __restrict__ B,
    int Npad, int K, int lda, int ldb,
    float* __restrict__ out0, int ld0,
    float* __restrict__ out1, int ld1,
    const float* __restrict__ bias,
    const float* __restrict__ memf,
    const float* __restrict__ reach,
    const float* __restrict__ xf32,
    int Nreal)
{
  const int lane = threadIdx.x & 31;
  const int wave = threadIdx.x >> 5;
  const int hi   = lane >> 4;
  const int l16  = lane & 15;

  const int mBase = blockIdx.y * 256 + wave * 32;
  const int nBase = blockIdx.x * 64;

  const __bf16* __restrict__ ar0 = A + (size_t)(mBase + l16) * lda;
  const __bf16* __restrict__ ar1 = ar0 + (size_t)16 * lda;
  const __bf16* __restrict__ br0 = B + (size_t)(nBase + l16) * ldb;
  const __bf16* __restrict__ br1 = br0 + (size_t)16 * ldb;
  const __bf16* __restrict__ br2 = br0 + (size_t)32 * ldb;
  const __bf16* __restrict__ br3 = br0 + (size_t)48 * ldb;

  v8f acc00 = {}, acc01 = {}, acc02 = {}, acc03 = {};
  v8f acc10 = {}, acc11 = {}, acc12 = {}, acc13 = {};

  for (int k0 = 0; k0 < K; k0 += 32) {
    __builtin_prefetch((const void*)(ar0 + k0 + 128), 0, 0);   // global_prefetch_b8
    __builtin_prefetch((const void*)(ar1 + k0 + 128), 0, 0);
    v16bf a0 = frag_ld(ar0, k0, hi);
    v16bf a1 = frag_ld(ar1, k0, hi);
    v16bf b0 = frag_ld(br0, k0, hi);
    v16bf b1 = frag_ld(br1, k0, hi);
    v16bf b2 = frag_ld(br2, k0, hi);
    v16bf b3 = frag_ld(br3, k0, hi);
    acc00 = __builtin_amdgcn_wmma_f32_16x16x32_bf16(false, a0, false, b0, (short)0, acc00, false, false);
    acc01 = __builtin_amdgcn_wmma_f32_16x16x32_bf16(false, a0, false, b1, (short)0, acc01, false, false);
    acc02 = __builtin_amdgcn_wmma_f32_16x16x32_bf16(false, a0, false, b2, (short)0, acc02, false, false);
    acc03 = __builtin_amdgcn_wmma_f32_16x16x32_bf16(false, a0, false, b3, (short)0, acc03, false, false);
    acc10 = __builtin_amdgcn_wmma_f32_16x16x32_bf16(false, a1, false, b0, (short)0, acc10, false, false);
    acc11 = __builtin_amdgcn_wmma_f32_16x16x32_bf16(false, a1, false, b1, (short)0, acc11, false, false);
    acc12 = __builtin_amdgcn_wmma_f32_16x16x32_bf16(false, a1, false, b2, (short)0, acc12, false, false);
    acc13 = __builtin_amdgcn_wmma_f32_16x16x32_bf16(false, a1, false, b3, (short)0, acc13, false, false);
  }

  // C/D 16x16 f32 layout: lane<16 -> M = r, lane>=16 -> M = 8+r, N = l16.
  v8f accs[2][4] = {{acc00, acc01, acc02, acc03}, {acc10, acc11, acc12, acc13}};
  #pragma unroll
  for (int g = 0; g < 2; ++g) {
    const int mRow0 = mBase + g * 16 + hi * 8;
    #pragma unroll
    for (int j = 0; j < 4; ++j) {
      const int col = nBase + j * 16 + l16;
      #pragma unroll
      for (int r = 0; r < 8; ++r) {
        const int m = mRow0 + r;
        const float v = accs[g][j][r];
        if (EPI == 0) {
          out0[(size_t)m * ld0 + col] = v;
        } else if (EPI == 1) {
          const float s   = tanhf(v + bias[col]);
          const float inf = s * memf[(size_t)m * ld0 + col];
          out1[(size_t)m * ld1 + col] = inf;                                   // infused -> d_out
          out0[(size_t)m * ld0 + col] = reach[m] * (xf32[(size_t)m * ld0 + col] + inf); // xo
        } else { // EPI == 2
          if (col < Nreal) out0[(size_t)m * ld0 + col] = v;
        }
      }
    }
  }
}

// --------------------------- prep / epilogue kernels -----------------------
__global__ void k_f32_to_bf16(const float* __restrict__ in, __bf16* __restrict__ out, size_t n) {
  size_t i = (size_t)blockIdx.x * 256 + threadIdx.x;
  if (i < n) out[i] = (__bf16)in[i];
}

__global__ void k_copy_f32(const float* __restrict__ in, float* __restrict__ out, size_t n) {
  size_t i = (size_t)blockIdx.x * 256 + threadIdx.x;
  if (i < n) out[i] = in[i];
}

// out[rPad][cols] = (r < rowsReal) ? bf16(in[r][cols]) : 0
__global__ void k_pad_convert(const float* __restrict__ in, __bf16* __restrict__ out,
                              int rowsReal, int rowsPad, int cols) {
  size_t i = (size_t)blockIdx.x * 256 + threadIdx.x;
  size_t n = (size_t)rowsPad * cols;
  if (i >= n) return;
  int r = (int)(i / cols);
  int c = (int)(i % cols);
  out[i] = (r < rowsReal) ? (__bf16)in[(size_t)r * cols + c] : (__bf16)0.0f;
}

// out[f][rPad] = (r < rowsReal) ? bf16(in[r][f]) : 0   (centroids^T for GEMM3)
__global__ void k_transpose_bf16(const float* __restrict__ in, __bf16* __restrict__ out,
                                 int rowsReal, int rowsPad, int cols) {
  size_t i = (size_t)blockIdx.x * 256 + threadIdx.x;
  size_t n = (size_t)cols * rowsPad;
  if (i >= n) return;
  int f = (int)(i / rowsPad);
  int r = (int)(i % rowsPad);
  out[i] = (r < rowsReal) ? (__bf16)in[(size_t)r * cols + f] : (__bf16)0.0f;
}

// c2[r] = ||centroid_r||^2 (padded rows -> +big so min ignores them)
__global__ void k_c2(const float* __restrict__ cent, float* __restrict__ c2,
                     int rowsReal, int cols) {
  int r = blockIdx.x;
  float a = 0.f;
  if (r < rowsReal)
    for (int c = threadIdx.x; c < cols; c += 256) {
      float v = cent[(size_t)r * cols + c];
      a += v * v;
    }
  float ss = block_reduce_sum(a);
  if (threadIdx.x == 0) c2[r] = (r < rowsReal) ? ss : 3.0e38f;
}

// wcos row-normalize, scale by COS_SCALE, convert to bf16 (padded rows -> 0)
__global__ void k_wcos(const float* __restrict__ w, __bf16* __restrict__ out,
                       int rowsReal, int cols) {
  int r = blockIdx.x;
  float a = 0.f;
  if (r < rowsReal)
    for (int c = threadIdx.x; c < cols; c += 256) {
      float v = w[(size_t)r * cols + c];
      a += v * v;
    }
  float ss = block_reduce_sum(a);
  float sc = (r < rowsReal) ? COS_SCALE * rsqrtf(ss + 1e-30f) : 0.f;
  for (int c = threadIdx.x; c < cols; c += 256) {
    float v = (r < rowsReal) ? w[(size_t)r * cols + c] * sc : 0.f;
    out[(size_t)r * cols + c] = (__bf16)v;
  }
}

// reach[m] = REACH_SCALE / (sqrt(max(min_c(||x||^2 - 2 x.c + ||c||^2),0)) + eps)
__global__ void k_reach(const float* __restrict__ x, const float* __restrict__ dots,
                        const float* __restrict__ c2, float* __restrict__ reach) {
  int m = blockIdx.x;
  float a = 0.f;
  for (int f = threadIdx.x; f < FEAT; f += 256) {
    float v = x[(size_t)m * FEAT + f];
    a += v * v;
  }
  float x2 = block_reduce_sum(a);
  float mn = 3.0e38f;
  for (int c = threadIdx.x; c < NCLSP; c += 256) {
    float d = x2 - 2.f * dots[(size_t)m * NCLSP + c] + c2[c];
    mn = fminf(mn, d);
  }
  mn = block_reduce_min(mn);
  if (threadIdx.x == 0) {
    float md = sqrtf(fmaxf(mn, 0.f));
    reach[m] = REACH_SCALE / (md + EPS_R);
  }
}

// softmax over 1000 classes -> bf16 vm (padded cols -> 0)
__global__ void k_softmax(const float* __restrict__ logits, const float* __restrict__ bias,
                          __bf16* __restrict__ vm) {
  int m = blockIdx.x;
  float mx = -3.0e38f;
  for (int c = threadIdx.x; c < NCLS; c += 256)
    mx = fmaxf(mx, logits[(size_t)m * NCLSP + c] + bias[c]);
  mx = block_reduce_max(mx);
  float se = 0.f;
  for (int c = threadIdx.x; c < NCLS; c += 256)
    se += __expf(logits[(size_t)m * NCLSP + c] + bias[c] - mx);
  se = block_reduce_sum(se);
  float inv = 1.f / se;
  for (int c = threadIdx.x; c < NCLSP; c += 256) {
    float v = (c < NCLS) ? __expf(logits[(size_t)m * NCLSP + c] + bias[c] - mx) * inv : 0.f;
    vm[(size_t)m * NCLSP + c] = (__bf16)v;
  }
}

// ex = xo / (1 + ||xo||)  (CosNorm input simplification), write bf16
__global__ void k_exnorm(const float* __restrict__ xo, __bf16* __restrict__ ex) {
  int m = blockIdx.x;
  float a = 0.f;
  for (int f = threadIdx.x; f < FEAT; f += 256) {
    float v = xo[(size_t)m * FEAT + f];
    a += v * v;
  }
  float ss = block_reduce_sum(a);
  float inv = 1.f / (1.f + sqrtf(ss));
  for (int f = threadIdx.x; f < FEAT; f += 256)
    ex[(size_t)m * FEAT + f] = (__bf16)(xo[(size_t)m * FEAT + f] * inv);
}

// ------------------------------- launch ------------------------------------
extern "C" void kernel_launch(void* const* d_in, const int* in_sizes, int n_in,
                              void* d_out, int out_size, void* d_ws, size_t ws_size,
                              hipStream_t stream) {
  (void)in_sizes; (void)n_in; (void)out_size; (void)ws_size;

  const float* x    = (const float*)d_in[0];
  const float* cent = (const float*)d_in[1];
  const float* Wh   = (const float*)d_in[2];
  const float* bh   = (const float*)d_in[3];
  const float* Ws   = (const float*)d_in[4];
  const float* bs   = (const float*)d_in[5];
  const float* Wc   = (const float*)d_in[6];

  float* out        = (float*)d_out;
  float* out_logits = out;                                   // [8192 x 1000]
  float* out_direct = out + (size_t)BATCH * NCLS;            // [8192 x 2048]
  float* out_inf    = out_direct + (size_t)BATCH * FEAT;     // [8192 x 2048]

  char* wsp = (char*)d_ws;
  auto alloc = [&](size_t bytes) -> char* {
    char* p = wsp;
    wsp += (bytes + 255) & ~(size_t)255;
    return p;
  };
  __bf16* xb   = (__bf16*)alloc((size_t)BATCH * FEAT * 2);   // x bf16
  __bf16* crow = (__bf16*)alloc((size_t)NCLSP * FEAT * 2);   // centroids [1024][2048]
  __bf16* ct   = (__bf16*)alloc((size_t)FEAT * NCLSP * 2);   // centroids^T [2048][1024]
  __bf16* whb  = (__bf16*)alloc((size_t)NCLSP * FEAT * 2);   // W_hall padded
  __bf16* wsb  = (__bf16*)alloc((size_t)FEAT * FEAT * 2);    // W_sel
  __bf16* wcb  = (__bf16*)alloc((size_t)NCLSP * FEAT * 2);   // W_cos normalized*16
  float*  c2   = (float*)alloc((size_t)NCLSP * 4);
  float*  reach= (float*)alloc((size_t)BATCH * 4);
  float*  big1 = (float*)alloc((size_t)BATCH * FEAT * 4);    // dots -> hall -> xo
  float*  big2 = (float*)alloc((size_t)BATCH * FEAT * 4);    // memf -> ex(bf16)
  __bf16* vmb  = (__bf16*)alloc((size_t)BATCH * NCLSP * 2);  // softmax(vm) bf16

  float*  dots = big1;               // [B][1024]
  float*  hall = big1;               // [B][1024] (reuses dots after k_reach)
  float*  xo   = big1;               // [B][2048] (reuses after k_softmax)
  float*  memf = big2;               // [B][2048]
  __bf16* exb  = (__bf16*)big2;      // [B][2048] bf16 (reuses memf after GEMM4)

  const dim3 t256(256);
  const size_t nx = (size_t)BATCH * FEAT;

  // -------- stage inputs (convert / pad / normalize) --------
  k_f32_to_bf16 <<<dim3((nx + 255) / 256), t256, 0, stream>>>(x, xb, nx);
  k_pad_convert <<<dim3(((size_t)NCLSP * FEAT + 255) / 256), t256, 0, stream>>>(cent, crow, NCLS, NCLSP, FEAT);
  k_transpose_bf16<<<dim3(((size_t)FEAT * NCLSP + 255) / 256), t256, 0, stream>>>(cent, ct, NCLS, NCLSP, FEAT);
  k_pad_convert <<<dim3(((size_t)NCLSP * FEAT + 255) / 256), t256, 0, stream>>>(Wh, whb, NCLS, NCLSP, FEAT);
  k_f32_to_bf16 <<<dim3(((size_t)FEAT * FEAT + 255) / 256), t256, 0, stream>>>(Ws, wsb, (size_t)FEAT * FEAT);
  k_wcos        <<<dim3(NCLSP), t256, 0, stream>>>(Wc, wcb, NCLS, FEAT);
  k_c2          <<<dim3(NCLSP), t256, 0, stream>>>(cent, c2, NCLS, FEAT);
  k_copy_f32    <<<dim3((nx + 255) / 256), t256, 0, stream>>>(x, out_direct, nx);

  const dim3 gC(NCLSP / 64, BATCH / 256);   // N=1024 GEMMs
  const dim3 gF(FEAT  / 64, BATCH / 256);   // N=2048 GEMMs

  // GEMM1: dots = x @ centroids^T   [B x 1024]
  gemm_bf16_wmma<0><<<gC, t256, 0, stream>>>(xb, crow, NCLSP, FEAT, FEAT, FEAT,
      dots, NCLSP, nullptr, 0, nullptr, nullptr, nullptr, nullptr, NCLSP);
  k_reach<<<dim3(BATCH), t256, 0, stream>>>(x, dots, c2, reach);

  // GEMM2: hall = x @ W_hall^T      [B x 1024]
  gemm_bf16_wmma<0><<<gC, t256, 0, stream>>>(xb, whb, NCLSP, FEAT, FEAT, FEAT,
      hall, NCLSP, nullptr, 0, nullptr, nullptr, nullptr, nullptr, NCLSP);
  k_softmax<<<dim3(BATCH), t256, 0, stream>>>(hall, bh, vmb);

  // GEMM3: memf = vm @ centroids    [B x 2048]  (B-operand = centroids^T rows)
  gemm_bf16_wmma<0><<<gF, t256, 0, stream>>>(vmb, ct, FEAT, NCLSP, NCLSP, NCLSP,
      memf, FEAT, nullptr, 0, nullptr, nullptr, nullptr, nullptr, FEAT);

  // GEMM4: selector + fuse: infused -> d_out, xo -> ws
  gemm_bf16_wmma<1><<<gF, t256, 0, stream>>>(xb, wsb, FEAT, FEAT, FEAT, FEAT,
      xo, FEAT, out_inf, FEAT, bs, memf, reach, x, FEAT);

  // ex = xo / (1 + ||xo||) -> bf16
  k_exnorm<<<dim3(BATCH), t256, 0, stream>>>(xo, exb);

  // GEMM5: logits = ex @ (16 * W_cos/||W_cos||)^T  -> d_out [B x 1000]
  gemm_bf16_wmma<2><<<gC, t256, 0, stream>>>(exb, wcb, NCLSP, FEAT, FEAT, FEAT,
      out_logits, NCLS, nullptr, 0, nullptr, nullptr, nullptr, nullptr, NCLS);
}
